// MultiHeadAttention_31928786878646
// MI455X (gfx1250) — matrix operand order repaired and checked by reading, not compiled, not run on hardware
//
#include <hip/hip_runtime.h>
#include <hip/hip_bf16.h>
#include <math.h>

typedef __bf16 bf16_t;
typedef __attribute__((ext_vector_type(16))) __bf16 v16bf;
typedef __attribute__((ext_vector_type(8)))  __bf16 bf16x8;
typedef __attribute__((ext_vector_type(8)))  float   v8f;

#define D_MODEL   1024
#define NUM_HEADS 16
#define D_HEAD    64
#define SEQ       1024
#define BATCH     4

__device__ inline bf16_t f2bf(float f) {
  unsigned u = __builtin_bit_cast(unsigned, f);
  unsigned r = u + 0x7FFFu + ((u >> 16) & 1u);   // round to nearest even
  unsigned short h = (unsigned short)(r >> 16);
  return __builtin_bit_cast(bf16_t, h);
}

// ---------------------------------------------------------------- convert
__global__ void cvt_f32_bf16(const float* __restrict__ in,
                             bf16_t* __restrict__ out, int n) {
  int i = blockIdx.x * blockDim.x + threadIdx.x;
  if (i < n) out[i] = f2bf(in[i]);
}

// ---------------------------------------------------------------- GEMM
// C[M,N] = A[M,K] @ B[N,K]^T, bf16 in, f32 accum.
// outMode: 0 = fp32 row-major, 1 = bf16 row-major,
//          2 = bf16 transposed-per-head VT[b,h,d,t] (MHA dims hardcoded)
#define BM 128
#define BN 128
#define BK 32

__global__ void __launch_bounds__(256)
gemm_bf16_wmma(const bf16_t* __restrict__ A, const bf16_t* __restrict__ B,
               float* __restrict__ Cf, bf16_t* __restrict__ Cb,
               int M, int N, int K, int outMode)
{
  __shared__ bf16_t As[BM * BK];
  __shared__ bf16_t Bs[BN * BK];

  const int t    = threadIdx.x;
  const int lane = t & 31;
  const int wid  = t >> 5;
  const int half = (lane >> 4) & 1;
  const int l15  = lane & 15;

  const int rowBase = blockIdx.y * BM;
  const int colBase = blockIdx.x * BN;
  const int wm = (wid & 3) * 32;    // wave row offset in tile
  const int wn = (wid >> 2) * 64;   // wave col offset in tile

  v8f acc[2][4];
  for (int i = 0; i < 2; ++i)
    for (int j = 0; j < 4; ++j)
      for (int e = 0; e < 8; ++e) acc[i][j][e] = 0.0f;

  const int lrow = t >> 1;          // 0..127
  const int lcol = (t & 1) * 16;    // 0 or 16

  for (int k0 = 0; k0 < K; k0 += BK) {
    const bf16x8* gA = reinterpret_cast<const bf16x8*>(A + (long)(rowBase + lrow) * K + k0 + lcol);
    const bf16x8* gB = reinterpret_cast<const bf16x8*>(B + (long)(colBase + lrow) * K + k0 + lcol);
    bf16x8 a0 = gA[0], a1 = gA[1];
    bf16x8 b0 = gB[0], b1 = gB[1];
    __syncthreads();
    *reinterpret_cast<bf16x8*>(&As[lrow * BK + lcol])     = a0;
    *reinterpret_cast<bf16x8*>(&As[lrow * BK + lcol + 8]) = a1;
    *reinterpret_cast<bf16x8*>(&Bs[lrow * BK + lcol])     = b0;
    *reinterpret_cast<bf16x8*>(&Bs[lrow * BK + lcol + 8]) = b1;
    __syncthreads();
    if (k0 + BK < K) {
      __builtin_prefetch(A + (long)(rowBase + lrow) * K + k0 + BK + lcol, 0, 0);
      __builtin_prefetch(B + (long)(colBase + lrow) * K + k0 + BK + lcol, 0, 0);
    }

    // A fragments: lanes 0-15 -> K 0..15, lanes 16-31 -> K 16..31, row = lane&15
    v16bf af[2];
    #pragma unroll
    for (int im = 0; im < 2; ++im) {
      const int r = wm + im * 16 + l15;
      bf16x8 L = *reinterpret_cast<const bf16x8*>(&As[r * BK + half * 16]);
      bf16x8 H = *reinterpret_cast<const bf16x8*>(&As[r * BK + half * 16 + 8]);
      #pragma unroll
      for (int e = 0; e < 8; ++e) { af[im][e] = L[e]; af[im][8 + e] = H[e]; }
    }
    // B fragments: lanes 0-15 -> K 0..15, lanes 16-31 -> K 16..31, col = lane&15
    v16bf bfr[4];
    #pragma unroll
    for (int in = 0; in < 4; ++in) {
      const int n = wn + in * 16 + l15;
      bf16x8 L = *reinterpret_cast<const bf16x8*>(&Bs[n * BK + half * 16]);
      bf16x8 H = *reinterpret_cast<const bf16x8*>(&Bs[n * BK + half * 16 + 8]);
      #pragma unroll
      for (int e = 0; e < 8; ++e) { bfr[in][e] = L[e]; bfr[in][8 + e] = H[e]; }
    }
    #pragma unroll
    for (int im = 0; im < 2; ++im)
      #pragma unroll
      for (int in = 0; in < 4; ++in)
        acc[im][in] = __builtin_amdgcn_wmma_f32_16x16x32_bf16(
            false, af[im], false, bfr[in], (short)0, acc[im][in], false, false);
  }

  if (outMode == 2) {
    // VT[b,h,d,t]: the 8 accumulator rows per lane are 8 consecutive t values
    #pragma unroll
    for (int im = 0; im < 2; ++im)
      #pragma unroll
      for (int in = 0; in < 4; ++in) {
        const int col  = colBase + wn + in * 16 + l15;       // h*64 + d
        const int row0 = rowBase + wm + im * 16 + 8 * half;  // b*SEQ + t (t mult of 8)
        bf16x8 pack;
        #pragma unroll
        for (int r = 0; r < 8; ++r) pack[r] = f2bf(acc[im][in][r]);
        const int b_ = row0 >> 10;
        const int t_ = row0 & (SEQ - 1);
        const int h_ = col >> 6;
        const int d_ = col & (D_HEAD - 1);
        *reinterpret_cast<bf16x8*>(
            Cb + ((long)((b_ * NUM_HEADS + h_) * D_HEAD + d_)) * SEQ + t_) = pack;
      }
    return;
  }

  #pragma unroll
  for (int im = 0; im < 2; ++im)
    #pragma unroll
    for (int in = 0; in < 4; ++in) {
      const int col = colBase + wn + in * 16 + l15;
      #pragma unroll
      for (int r = 0; r < 8; ++r) {
        const int row = rowBase + wm + im * 16 + r + 8 * half;
        float v = acc[im][in][r];
        if (outMode == 1) Cb[(long)row * N + col] = f2bf(v);
        else              Cf[(long)row * N + col] = v;
      }
    }
}

// ---------------------------------------------------------------- RoPE
__global__ void rope_f32_to_bf16(const float* __restrict__ in,
                                 bf16_t* __restrict__ out, int total) {
  int id = blockIdx.x * blockDim.x + threadIdx.x;
  if (id >= total) return;
  const int i   = id & 31;          // pair index within head (0..31)
  const int hw  = id >> 5;
  const int h   = hw & (NUM_HEADS - 1);
  const int row = hw >> 4;          // b*T + t
  const int t   = row & (SEQ - 1);
  const int col = h * D_HEAD + 2 * i;
  float x1 = in[(long)row * D_MODEL + col];
  float x2 = in[(long)row * D_MODEL + col + 1];
  float freq = __expf(-((float)(2 * i) / (float)D_HEAD) * 9.210340371976184f); // ln(10000)
  float ang = (float)t * freq;
  float s, c;
  __sincosf(ang, &s, &c);
  out[(long)row * D_MODEL + col]     = f2bf(x1 * c - x2 * s);
  out[(long)row * D_MODEL + col + 1] = f2bf(x1 * s + x2 * c);
}

// ---------------------------------------------------------------- flash attention
// One wave per (b, h, 16-query tile). Causal, online softmax, WMMA bf16.
// K is row-major [b*T, D]; V is pre-transposed VT[b,h,d,t] so both QK^T and PV
// B-fragments are contiguous 16-byte global loads (L2-resident reuse).
__global__ void __launch_bounds__(32)
flash_attn_wmma(const bf16_t* __restrict__ Q, const bf16_t* __restrict__ K,
                const bf16_t* __restrict__ VT, bf16_t* __restrict__ O)
{
  __shared__ bf16_t Pt[16 * 32];

  const int lane = threadIdx.x & 31;
  const int half = lane >> 4;
  const int l15  = lane & 15;

  const int qBase = blockIdx.x * 16;
  const int h = blockIdx.y;
  const int b = blockIdx.z;
  const long headOff = ((long)b * SEQ) * D_MODEL + (long)h * D_HEAD;
  const bf16_t* Qh  = Q + headOff;
  const bf16_t* Kh  = K + headOff;
  const bf16_t* VTh = VT + (long)(b * NUM_HEADS + h) * D_HEAD * SEQ;
  bf16_t* Oh = O + headOff;

  // Q fragments (16x64 as two 16x32 A-frags), loaded once
  v16bf qa[2];
  #pragma unroll
  for (int dc = 0; dc < 2; ++dc) {
    const int row = qBase + l15;
    bf16x8 L = *reinterpret_cast<const bf16x8*>(Qh + (long)row * D_MODEL + dc * 32 + half * 16);
    bf16x8 H = *reinterpret_cast<const bf16x8*>(Qh + (long)row * D_MODEL + dc * 32 + half * 16 + 8);
    #pragma unroll
    for (int e = 0; e < 8; ++e) { qa[dc][e] = L[e]; qa[dc][8 + e] = H[e]; }
  }

  v8f o[4];
  for (int f = 0; f < 4; ++f) for (int e = 0; e < 8; ++e) o[f][e] = 0.0f;
  float m_run[8], l_run[8];
  for (int r = 0; r < 8; ++r) { m_run[r] = -1e30f; l_run[r] = 0.0f; }
  const float scale = 0.125f;  // 1/sqrt(64)

  for (int kBase = 0; kBase < qBase + 16; kBase += 32) {
    // S = Q @ K^T for two 16-key fragments; K B-frags straight from global
    v8f s[2];
    for (int nf = 0; nf < 2; ++nf) for (int e = 0; e < 8; ++e) s[nf][e] = 0.0f;
    #pragma unroll
    for (int nf = 0; nf < 2; ++nf) {
      #pragma unroll
      for (int dc = 0; dc < 2; ++dc) {
        v16bf kb;
        const bf16_t* kp = Kh + (long)(kBase + nf * 16 + l15) * D_MODEL + dc * 32 + half * 16;
        bf16x8 L = *reinterpret_cast<const bf16x8*>(kp);
        bf16x8 H = *reinterpret_cast<const bf16x8*>(kp + 8);
        #pragma unroll
        for (int e = 0; e < 8; ++e) { kb[e] = L[e]; kb[8 + e] = H[e]; }
        s[nf] = __builtin_amdgcn_wmma_f32_16x16x32_bf16(
            false, qa[dc], false, kb, (short)0, s[nf], false, false);
      }
    }
    if (kBase + 32 < qBase + 16)
      __builtin_prefetch(Kh + (long)(kBase + 32 + l15) * D_MODEL + half * 16, 0, 0);

    // online softmax; row r lives at (r + 8*half), cols across 16-lane half
    float alpha[8];
    #pragma unroll
    for (int r = 0; r < 8; ++r) {
      const int qRow = qBase + r + 8 * half;
      float s0 = ((kBase + l15)      <= qRow) ? s[0][r] * scale : -1e30f;
      float s1 = ((kBase + 16 + l15) <= qRow) ? s[1][r] * scale : -1e30f;
      float sm = fmaxf(s0, s1);
      #pragma unroll
      for (int msk = 1; msk < 16; msk <<= 1) sm = fmaxf(sm, __shfl_xor(sm, msk, 32));
      float mNew = fmaxf(m_run[r], sm);
      float a  = __expf(m_run[r] - mNew);
      float p0 = __expf(s0 - mNew);
      float p1 = __expf(s1 - mNew);
      float rs = p0 + p1;
      #pragma unroll
      for (int msk = 1; msk < 16; msk <<= 1) rs += __shfl_xor(rs, msk, 32);
      l_run[r] = l_run[r] * a + rs;
      m_run[r] = mNew;
      alpha[r] = a;
      Pt[(r + 8 * half) * 32 + l15]      = f2bf(p0);
      Pt[(r + 8 * half) * 32 + 16 + l15] = f2bf(p1);
    }
    __syncthreads();

    #pragma unroll
    for (int f = 0; f < 4; ++f)
      #pragma unroll
      for (int r = 0; r < 8; ++r) o[f][r] *= alpha[r];

    // reload P in A-fragment layout (LDS transpose)
    v16bf pa;
    {
      bf16x8 L = *reinterpret_cast<const bf16x8*>(&Pt[l15 * 32 + half * 16]);
      bf16x8 H = *reinterpret_cast<const bf16x8*>(&Pt[l15 * 32 + half * 16 + 8]);
      #pragma unroll
      for (int e = 0; e < 8; ++e) { pa[e] = L[e]; pa[8 + e] = H[e]; }
    }

    // O += P @ V ; V B-frags contiguous from VT[b,h,d,t]
    #pragma unroll
    for (int f = 0; f < 4; ++f) {
      v16bf vb;
      const bf16_t* vp = VTh + (long)(f * 16 + l15) * SEQ + kBase + half * 16;
      bf16x8 L = *reinterpret_cast<const bf16x8*>(vp);
      bf16x8 H = *reinterpret_cast<const bf16x8*>(vp + 8);
      #pragma unroll
      for (int e = 0; e < 8; ++e) { vb[e] = L[e]; vb[8 + e] = H[e]; }
      o[f] = __builtin_amdgcn_wmma_f32_16x16x32_bf16(
          false, pa, false, vb, (short)0, o[f], false, false);
    }
    __syncthreads();
  }

  #pragma unroll
  for (int f = 0; f < 4; ++f) {
    const int d = f * 16 + l15;
    #pragma unroll
    for (int r = 0; r < 8; ++r) {
      const int row = qBase + r + 8 * half;
      Oh[(long)row * D_MODEL + d] = f2bf(o[f][r] / l_run[r]);
    }
  }
}

// ---------------------------------------------------------------- launch
extern "C" void kernel_launch(void* const* d_in, const int* in_sizes, int n_in,
                              void* d_out, int out_size, void* d_ws, size_t ws_size,
                              hipStream_t stream)
{
  const float* x  = (const float*)d_in[0];
  const float* Wq = (const float*)d_in[1];
  const float* Wk = (const float*)d_in[2];
  const float* Wv = (const float*)d_in[3];
  const float* Wo = (const float*)d_in[4];
  float* out = (float*)d_out;

  const int BT = BATCH * SEQ;   // 4096
  const int D  = D_MODEL;

  char* ws = (char*)d_ws;
  size_t off = 0;
  auto alloc = [&](size_t bytes) -> void* {
    void* p = ws + off;
    off += (bytes + 255) & ~(size_t)255;
    return p;
  };

  bf16_t* xb  = (bf16_t*)alloc((size_t)BT * D * 2);
  bf16_t* wqb = (bf16_t*)alloc((size_t)D * D * 2);
  bf16_t* wkb = (bf16_t*)alloc((size_t)D * D * 2);
  bf16_t* wvb = (bf16_t*)alloc((size_t)D * D * 2);
  bf16_t* wob = (bf16_t*)alloc((size_t)D * D * 2);
  float*  Qf  = (float*) alloc((size_t)BT * D * 4);
  float*  Kf  = (float*) alloc((size_t)BT * D * 4);
  bf16_t* Qb  = (bf16_t*)alloc((size_t)BT * D * 2);
  bf16_t* Kb  = (bf16_t*)alloc((size_t)BT * D * 2);
  bf16_t* VTb = (bf16_t*)alloc((size_t)BT * D * 2);  // VT[b,h,d,t]
  bf16_t* Cb  = (bf16_t*)alloc((size_t)BT * D * 2);

  const int nX = BT * D;
  const int nW = D * D;
  cvt_f32_bf16<<<(nX + 255) / 256, 256, 0, stream>>>(x,  xb,  nX);
  cvt_f32_bf16<<<(nW + 255) / 256, 256, 0, stream>>>(Wq, wqb, nW);
  cvt_f32_bf16<<<(nW + 255) / 256, 256, 0, stream>>>(Wk, wkb, nW);
  cvt_f32_bf16<<<(nW + 255) / 256, 256, 0, stream>>>(Wv, wvb, nW);
  cvt_f32_bf16<<<(nW + 255) / 256, 256, 0, stream>>>(Wo, wob, nW);

  dim3 gG(D / BN, BT / BM);   // (8, 32)
  gemm_bf16_wmma<<<gG, 256, 0, stream>>>(xb, wqb, Qf, nullptr, BT, D, D, 0);
  gemm_bf16_wmma<<<gG, 256, 0, stream>>>(xb, wkb, Kf, nullptr, BT, D, D, 0);
  gemm_bf16_wmma<<<gG, 256, 0, stream>>>(xb, wvb, nullptr, VTb, BT, D, D, 2);

  const int nPairs = BT * NUM_HEADS * (D_HEAD / 2);  // 2,097,152
  rope_f32_to_bf16<<<(nPairs + 255) / 256, 256, 0, stream>>>(Qf, Qb, nPairs);
  rope_f32_to_bf16<<<(nPairs + 255) / 256, 256, 0, stream>>>(Kf, Kb, nPairs);

  dim3 gF(SEQ / 16, NUM_HEADS, BATCH);
  flash_attn_wmma<<<gF, 32, 0, stream>>>(Qb, Kb, VTb, Cb);

  gemm_bf16_wmma<<<gG, 256, 0, stream>>>(Cb, wob, out, nullptr, BT, D, D, 0);
}
